// DAWN_84035330113861
// MI455X (gfx1250) — matrix operand order repaired and checked by reading, not compiled, not run on hardware
//
#include <hip/hip_runtime.h>
#include <hip/hip_bf16.h>

typedef __bf16 bf16;
typedef __attribute__((ext_vector_type(16))) __bf16 bf16x16;
typedef __attribute__((ext_vector_type(8)))  float  f32x8;
typedef unsigned int u32x4 __attribute__((ext_vector_type(4)));
typedef int          i32x4 __attribute__((ext_vector_type(4)));
typedef int          i32x8 __attribute__((ext_vector_type(8)));

#define NB      2
#define NS      2048
#define ND      1024
#define NHEADS  16
#define DHEAD   64
#define DBR     128
#define N_QK    4096
#define N_KNOW  8192
#define MROWS   4096   // NB*NS

// ---------------------------------------------------------------------------
// WMMA fragment loaders (CDNA5 wave32 layouts, 16-bit elements, K-step = 32)
// ---------------------------------------------------------------------------
// A (16x32, row-major [16 x lda]): lane L holds row L%16;
//   elements 0..7  = K: k0 + (L/16)*8 .. +7
//   elements 8..15 = K: k0 + 16 + (L/16)*8 .. +7
__device__ __forceinline__ bf16x16 load_a_frag(const bf16* __restrict__ A, int lda, int k0) {
    int lane = threadIdx.x & 31;
    const bf16* p = A + (size_t)(lane & 15) * lda + k0 + ((lane >> 4) << 3);
    union { bf16x16 v; uint4 u[2]; } f;
    f.u[0] = *(const uint4*)(p);
    f.u[1] = *(const uint4*)(p + 16);
    return f.v;
}

// B via B_T (row-major [N x ldb], B_T[n][k] = B[k][n]): lane L holds col L%16;
//   elements 0..7  = K: k0 + (L/16)*8 .. +7
//   elements 8..15 = K: k0 + 16 + (L/16)*8 .. +7
__device__ __forceinline__ bf16x16 load_b_frag(const bf16* __restrict__ BT, int ldb, int k0) {
    int lane = threadIdx.x & 31;
    const bf16* p = BT + (size_t)(lane & 15) * ldb + k0 + ((lane >> 4) << 3);
    union { bf16x16 v; uint4 u[2]; } f;
    f.u[0] = *(const uint4*)(p);
    f.u[1] = *(const uint4*)(p + 16);
    return f.v;
}

#define WMMA_BF16(A_, B_, C_) \
    __builtin_amdgcn_wmma_f32_16x16x32_bf16(false, (A_), false, (B_), (short)0, (C_), false, false)

// ---------------------------------------------------------------------------
// TDM: 2-D tile load (Global -> LDS) via Tensor Data Mover descriptor (D#).
// Loads tile_n rows x tile_k cols of bf16 (row stride ld_k elems) into LDS,
// packed row-major [tile_n][tile_k].
// D# layout per CDNA5 ISA ch.8: group0 = {flags, lds_addr, global_addr, type=2},
// group1 = {mask/data_size, dims, tile dims, strides}. Groups 2/3 zero (2-D).
// This toolchain exposes the 6-arg builtin (extra i32x8 group, then cpol).
// ---------------------------------------------------------------------------
__device__ __forceinline__ void tdm_load_2d(unsigned lds_off, const void* gaddr,
                                            unsigned tile_k, unsigned tile_n,
                                            unsigned ld_k) {
    unsigned long long ga = (unsigned long long)(uintptr_t)gaddr;
    u32x4 g0;
    g0[0] = 1u;                                           // count=1, load, user mode
    g0[1] = lds_off;                                      // lds_addr (bytes)
    g0[2] = (unsigned)(ga & 0xffffffffu);                 // global_addr[31:0]
    g0[3] = (unsigned)((ga >> 32) & 0x01ffffffu) | (2u << 30); // addr[56:32] | type=2
    unsigned td0 = ld_k;                                  // tensor_dim0 (elems per row)
    unsigned td1 = tile_n;                                // tensor_dim1 (rows from tile start)
    i32x8 g1;
    g1[0] = (int)(1u << 16);                              // data_size=1 -> 2 bytes
    g1[1] = (int)((td0 & 0xffffu) << 16);                 // tensor_dim0[15:0]
    g1[2] = (int)(((td0 >> 16) & 0xffffu) | ((td1 & 0xffffu) << 16));
    g1[3] = (int)(((td1 >> 16) & 0xffffu) | ((tile_k & 0xffffu) << 16)); // tile_dim0
    g1[4] = (int)(tile_n & 0xffffu);                      // tile_dim1 (tile_dim2=0)
    g1[5] = (int)ld_k;                                    // tensor_dim0_stride[31:0]
    g1[6] = 0;
    g1[7] = 0;
    i32x4 z4 = {0, 0, 0, 0};
    i32x8 z8 = {0, 0, 0, 0, 0, 0, 0, 0};
    __builtin_amdgcn_tensor_load_to_lds(g0, g1, z4, z4, z8, 0);
}

// ---------------------------------------------------------------------------
// GEMM: C[M,N] = A_bf16[M,K] @ B (given as B_T[N,K] bf16) (+bias)(+residual)
// Workgroup (256 thr = 8 waves) computes a 128x128 tile. B panel (128 x 64K)
// is staged into LDS by the TDM (wave 0 issues, TENSORcnt double-buffered);
// each wave owns 16 rows of A (direct global loads) x 128 cols from LDS.
// B-fragment stream is software-pipelined one step ahead so ds_load latency
// overlaps wmma issue. Requires M%128==0, N%128==0, K%64==0.
// ---------------------------------------------------------------------------
__global__ void gemm_bf16_wmma(const bf16* __restrict__ A, const bf16* __restrict__ BT,
                               float* __restrict__ C, int M, int N, int K,
                               const float* __restrict__ bias,
                               const float* __restrict__ residual) {
    __shared__ __align__(32) bf16 ldsB[2][128 * 64];
    int tiles_n = N >> 7;
    int m0 = ((int)blockIdx.x / tiles_n) << 7;
    int n0 = ((int)blockIdx.x % tiles_n) << 7;
    int wid = threadIdx.x >> 5;
    int mw  = m0 + (wid << 4);                  // this wave's 16 rows
    const bf16* Ap = A + (size_t)mw * K;
    const bf16* Bp = BT + (size_t)n0 * K;

    f32x8 acc[8] = {};
    int nkb = K >> 6;
    bool issuer = (threadIdx.x < 32);           // wave 0 drives the TDM
    if (issuer)
        tdm_load_2d((unsigned)(uintptr_t)&ldsB[0][0], Bp, 64, 128, (unsigned)K);

    for (int j = 0; j < nkb; ++j) {
        if (issuer) {
            if (j + 1 < nkb) {
                tdm_load_2d((unsigned)(uintptr_t)&ldsB[(j + 1) & 1][0],
                            Bp + (size_t)(j + 1) * 64, 64, 128, (unsigned)K);
                __builtin_amdgcn_s_wait_tensorcnt(1);   // current buffer ready
            } else {
                __builtin_amdgcn_s_wait_tensorcnt(0);
            }
        }
        __syncthreads();                                 // panel visible to all waves
        if (j + 1 < nkb)
            __builtin_prefetch(Ap + (size_t)(j + 1) * 64, 0, 1);
        const bf16* Bl = &ldsB[j & 1][0];
#pragma unroll
        for (int ks = 0; ks < 2; ++ks) {
            bf16x16 a = load_a_frag(Ap, K, j * 64 + ks * 32);
            // software-pipelined B stream: fragment t+1 in flight while wmma t issues
            bf16x16 bcur = load_b_frag(Bl, 64, ks * 32);
#pragma unroll
            for (int t = 0; t < 8; ++t) {
                bf16x16 bnext;
                if (t < 7)
                    bnext = load_b_frag(Bl + (size_t)((t + 1) << 4) * 64, 64, ks * 32);
                acc[t] = WMMA_BF16(a, bcur, acc[t]);
                if (t < 7) bcur = bnext;
            }
        }
        __syncthreads();                                 // reads done before overwrite
    }

    int lane = threadIdx.x & 31;
    int colo = lane & 15;
    int rhi  = (lane >> 4) << 3;
#pragma unroll
    for (int t = 0; t < 8; ++t) {
        int n = n0 + (t << 4) + colo;
        float bv = bias ? bias[n] : 0.0f;
#pragma unroll
        for (int r = 0; r < 8; ++r) {
            int m = mw + r + rhi;
            float v = acc[t][r] + bv;
            if (residual) v += residual[(size_t)m * N + n];
            C[(size_t)m * N + n] = v;
        }
    }
}

// ---------------------------------------------------------------------------
// Flash attention: 1 wave per (b, h, 16-row Q tile). KV blocks of 32.
// Qh,Kh: bf16 [B,H,S,64] row-major; VT: bf16 [B,H,64,S]; O: f32 [B,S,D].
// ---------------------------------------------------------------------------
__global__ void flash_attn_wmma(const bf16* __restrict__ Qh, const bf16* __restrict__ Kh,
                                const bf16* __restrict__ VT, float* __restrict__ O) {
    __shared__ bf16 lds_p[8 * 16 * 32];
    int wid  = threadIdx.x >> 5;
    int wave = (int)(blockIdx.x * (blockDim.x >> 5) + wid);
    const int QT = NS / 16;
    int b  = wave / (NHEADS * QT);
    int r0 = wave % (NHEADS * QT);
    int h  = r0 / QT;
    int qt = r0 % QT;
    int q0 = qt << 4;

    const bf16* Qp = Qh + (size_t)(b * NHEADS + h) * NS * DHEAD;
    const bf16* Kp = Kh + (size_t)(b * NHEADS + h) * NS * DHEAD;
    const bf16* Vp = VT + (size_t)(b * NHEADS + h) * DHEAD * NS;
    bf16* myp = lds_p + wid * (16 * 32);

    int lane = threadIdx.x & 31;
    int colo = lane & 15;
    int rhi  = (lane >> 4) << 3;

    bf16x16 aq0 = load_a_frag(Qp + (size_t)q0 * DHEAD, DHEAD, 0);
    bf16x16 aq1 = load_a_frag(Qp + (size_t)q0 * DHEAD, DHEAD, 32);

    f32x8 o0 = {}, o1 = {}, o2 = {}, o3 = {};
    float mrun[8], lrun[8];
#pragma unroll
    for (int r = 0; r < 8; ++r) { mrun[r] = -1e30f; lrun[r] = 0.0f; }

    const float rscale = 0.125f;                   // 1/sqrt(64)
    int nkv = (q0 + 15) / 32 + 1;                  // causal block bound
    for (int j = 0; j < nkv; ++j) {
        int kvb = j << 5;
        f32x8 s0 = {}, s1 = {};
        {
            bf16x16 bk;
            bk = load_b_frag(Kp + (size_t)(kvb     ) * DHEAD, DHEAD, 0);
            s0 = WMMA_BF16(aq0, bk, s0);
            bk = load_b_frag(Kp + (size_t)(kvb     ) * DHEAD, DHEAD, 32);
            s0 = WMMA_BF16(aq1, bk, s0);
            bk = load_b_frag(Kp + (size_t)(kvb + 16) * DHEAD, DHEAD, 0);
            s1 = WMMA_BF16(aq0, bk, s1);
            bk = load_b_frag(Kp + (size_t)(kvb + 16) * DHEAD, DHEAD, 32);
            s1 = WMMA_BF16(aq1, bk, s1);
        }
#pragma unroll
        for (int r = 0; r < 8; ++r) {
            int row = q0 + r + rhi;
            s0[r] = ((kvb + colo)      <= row) ? s0[r] * rscale : -1e30f;
            s1[r] = ((kvb + 16 + colo) <= row) ? s1[r] * rscale : -1e30f;
        }
#pragma unroll
        for (int r = 0; r < 8; ++r) {
            float mx = fmaxf(s0[r], s1[r]);
            mx = fmaxf(mx, __shfl_xor(mx, 1, 32));
            mx = fmaxf(mx, __shfl_xor(mx, 2, 32));
            mx = fmaxf(mx, __shfl_xor(mx, 4, 32));
            mx = fmaxf(mx, __shfl_xor(mx, 8, 32));
            float mnew = fmaxf(mrun[r], mx);
            float corr = __expf(mrun[r] - mnew);
            float p0 = __expf(s0[r] - mnew);
            float p1 = __expf(s1[r] - mnew);
            float rs = p0 + p1;
            rs += __shfl_xor(rs, 1, 32);
            rs += __shfl_xor(rs, 2, 32);
            rs += __shfl_xor(rs, 4, 32);
            rs += __shfl_xor(rs, 8, 32);
            lrun[r] = lrun[r] * corr + rs;
            mrun[r] = mnew;
            o0[r] *= corr; o1[r] *= corr; o2[r] *= corr; o3[r] *= corr;
            myp[(r + rhi) * 32 + colo]      = (bf16)p0;
            myp[(r + rhi) * 32 + 16 + colo] = (bf16)p1;
        }
        asm volatile("s_wait_dscnt 0" ::: "memory");
        bf16x16 ap = load_a_frag(myp, 32, 0);
        asm volatile("s_wait_dscnt 0" ::: "memory");
        {
            bf16x16 bv;
            bv = load_b_frag(Vp + (size_t)0  * NS, NS, kvb); o0 = WMMA_BF16(ap, bv, o0);
            bv = load_b_frag(Vp + (size_t)16 * NS, NS, kvb); o1 = WMMA_BF16(ap, bv, o1);
            bv = load_b_frag(Vp + (size_t)32 * NS, NS, kvb); o2 = WMMA_BF16(ap, bv, o2);
            bv = load_b_frag(Vp + (size_t)48 * NS, NS, kvb); o3 = WMMA_BF16(ap, bv, o3);
        }
    }
#pragma unroll
    for (int r = 0; r < 8; ++r) {
        float inv = 1.0f / (lrun[r] + 1e-20f);
        int row = q0 + r + rhi;
        float* orow = O + ((size_t)b * NS + row) * ND + h * DHEAD;
        orow[colo]      = o0[r] * inv;
        orow[16 + colo] = o1[r] * inv;
        orow[32 + colo] = o2[r] * inv;
        orow[48 + colo] = o3[r] * inv;
    }
}

// ---------------------------------------------------------------------------
// Elementwise / reduction helpers
// ---------------------------------------------------------------------------
__global__ void layernorm_kernel(const float* __restrict__ x, const float* __restrict__ sc,
                                 const float* __restrict__ bi, float* __restrict__ of,
                                 bf16* __restrict__ ob) {
    int row = blockIdx.x;
    const float* xr = x + (size_t)row * ND;
    float s = 0.f, s2 = 0.f;
    for (int i = threadIdx.x; i < ND; i += blockDim.x) { float v = xr[i]; s += v; s2 += v * v; }
    __shared__ float sh0[8], sh1[8];
    for (int m = 16; m >= 1; m >>= 1) { s += __shfl_xor(s, m, 32); s2 += __shfl_xor(s2, m, 32); }
    if ((threadIdx.x & 31) == 0) { sh0[threadIdx.x >> 5] = s; sh1[threadIdx.x >> 5] = s2; }
    __syncthreads();
    float ts = 0.f, ts2 = 0.f;
    for (int w = 0; w < 8; ++w) { ts += sh0[w]; ts2 += sh1[w]; }
    float mean = ts / ND;
    float var  = ts2 / ND - mean * mean;
    float rstd = rsqrtf(var + 1e-6f);
    for (int i = threadIdx.x; i < ND; i += blockDim.x) {
        float v = (xr[i] - mean) * rstd * sc[i] + bi[i];
        of[(size_t)row * ND + i] = v;
        ob[(size_t)row * ND + i] = (bf16)v;
    }
}

__global__ void rownorm_kernel(const float* __restrict__ e, bf16* __restrict__ o, int cols) {
    int row = blockIdx.x;
    const float* er = e + (size_t)row * cols;
    float s2 = 0.f;
    for (int i = threadIdx.x; i < cols; i += blockDim.x) { float v = er[i]; s2 += v * v; }
    __shared__ float sh[8];
    for (int m = 16; m >= 1; m >>= 1) s2 += __shfl_xor(s2, m, 32);
    if ((threadIdx.x & 31) == 0) sh[threadIdx.x >> 5] = s2;
    __syncthreads();
    float t = 0.f;
    int nw = blockDim.x >> 5;
    for (int w = 0; w < nw; ++w) t += sh[w];
    float inv = 1.0f / (sqrtf(t) + 1e-8f);
    for (int i = threadIdx.x; i < cols; i += blockDim.x)
        o[(size_t)row * cols + i] = (bf16)(er[i] * inv);
}

__global__ void transpose_bf16_kernel(const float* __restrict__ in, bf16* __restrict__ out,
                                      int R, int C) {
    size_t i = blockIdx.x * (size_t)blockDim.x + threadIdx.x;
    if (i >= (size_t)R * C) return;
    int r = (int)(i / C), c = (int)(i % C);
    out[(size_t)c * R + r] = (bf16)in[i];
}

__global__ void slice_bf16_kernel(const float* __restrict__ in, bf16* __restrict__ out,
                                  int rows, int src_ld, int off, int cols) {
    size_t i = blockIdx.x * (size_t)blockDim.x + threadIdx.x;
    if (i >= (size_t)rows * cols) return;
    int r = (int)(i / cols), c = (int)(i % cols);
    out[i] = (bf16)in[(size_t)r * src_ld + off + c];
}

__global__ void tau_kernel(const float* __restrict__ normed, const float* __restrict__ tk,
                           const float* __restrict__ tb, float* __restrict__ tau, int ncols) {
    int row = blockIdx.x;
    const float* nr = normed + (size_t)row * ND;
    float acc[3] = {0.f, 0.f, 0.f};
    for (int i = threadIdx.x; i < ND; i += blockDim.x) {
        float v = nr[i];
        for (int j = 0; j < ncols; ++j) acc[j] += v * tk[i * ncols + j];
    }
    __shared__ float sh[3][8];
    for (int j = 0; j < ncols; ++j) {
        float a = acc[j];
        for (int m = 16; m >= 1; m >>= 1) a += __shfl_xor(a, m, 32);
        acc[j] = a;
    }
    if ((threadIdx.x & 31) == 0)
        for (int j = 0; j < ncols; ++j) sh[j][threadIdx.x >> 5] = acc[j];
    __syncthreads();
    if (threadIdx.x == 0) {
        int nw = blockDim.x >> 5;
        for (int j = 0; j < ncols; ++j) {
            float t = 0.f;
            for (int w = 0; w < nw; ++w) t += sh[j][w];
            tau[(size_t)row * ncols + j] = t + tb[j];
        }
    }
}

__global__ void gate_norm_kernel(const float* __restrict__ sraw, const float* __restrict__ tau,
                                 int tstride, bf16* __restrict__ g, int N) {
    int row = blockIdx.x;
    float tv = tau[(size_t)row * tstride];
    const float* sr = sraw + (size_t)row * N;
    float s = 0.f;
    for (int i = threadIdx.x; i < N; i += blockDim.x) {
        float v = sr[i] - tv;
        s += (v > 0.f) ? v : 0.f;
    }
    __shared__ float sh[8];
    for (int m = 16; m >= 1; m >>= 1) s += __shfl_xor(s, m, 32);
    if ((threadIdx.x & 31) == 0) sh[threadIdx.x >> 5] = s;
    __syncthreads();
    float tot = 0.f;
    for (int w = 0; w < 8; ++w) tot += sh[w];
    float inv = 1.0f / (tot + 1e-8f);
    for (int i = threadIdx.x; i < N; i += blockDim.x) {
        float v = sr[i] - tv;
        v = (v > 0.f) ? v : 0.f;
        g[(size_t)row * N + i] = (bf16)(v * inv);
    }
}

__global__ void colsum_kernel(const bf16* __restrict__ g, float* __restrict__ cs, int M, int N) {
    int c = (int)(blockIdx.x * blockDim.x + threadIdx.x);
    if (c >= N) return;
    float s = 0.f;
    for (int r = 0; r < M; ++r) s += (float)g[(size_t)r * N + c];
    cs[c] = s;
}

__global__ void aux_kernel(const float* __restrict__ cs, int N, float tgt, float* __restrict__ auxp) {
    float s = 0.f;
    for (int i = threadIdx.x; i < N; i += blockDim.x) {
        float d = cs[i] * (1.0f / (float)MROWS) - tgt;
        s += d * d;
    }
    __shared__ float sh[8];
    for (int m = 16; m >= 1; m >>= 1) s += __shfl_xor(s, m, 32);
    if ((threadIdx.x & 31) == 0) sh[threadIdx.x >> 5] = s;
    __syncthreads();
    if (threadIdx.x == 0) {
        float t = 0.f;
        for (int w = 0; w < 8; ++w) t += sh[w];
        atomicAdd(auxp, t * (float)N);
    }
}

__global__ void head_prep_kernel(const float* __restrict__ in, bf16* __restrict__ out, int do_t) {
    size_t i = blockIdx.x * (size_t)blockDim.x + threadIdx.x;
    if (i >= (size_t)NB * NS * ND) return;
    int d  = (int)(i % ND);
    size_t bs = i / ND;
    int s_ = (int)(bs % NS);
    int b  = (int)(bs / NS);
    int h  = d / DHEAD, dd = d % DHEAD;
    bf16 v = (bf16)in[i];
    if (!do_t) out[((size_t)(b * NHEADS + h) * NS + s_) * DHEAD + dd] = v;
    else       out[((size_t)(b * NHEADS + h) * DHEAD + dd) * NS + s_] = v;
}

__global__ void zerof_kernel(float* __restrict__ p, size_t n) {
    size_t i = blockIdx.x * (size_t)blockDim.x + threadIdx.x;
    if (i < n) p[i] = 0.f;
}

// ---------------------------------------------------------------------------
extern "C" void kernel_launch(void* const* d_in, const int* in_sizes, int n_in,
                              void* d_out, int out_size, void* d_ws, size_t ws_size,
                              hipStream_t stream) {
    (void)in_sizes; (void)n_in; (void)ws_size;
    const float* x          = (const float*)d_in[0];
    const float* qk_emb     = (const float*)d_in[1];
    const float* v_emb      = (const float*)d_in[2];
    const float* know_emb   = (const float*)d_in[3];
    const float* qk_w       = (const float*)d_in[4];
    const float* v_w        = (const float*)d_in[5];
    const float* know_w     = (const float*)d_in[6];
    const float* proj_attn_k = (const float*)d_in[7];
    const float* proj_attn_b = (const float*)d_in[8];
    const float* tau_attn_k  = (const float*)d_in[9];
    const float* tau_attn_b  = (const float*)d_in[10];
    const float* proj_know_k = (const float*)d_in[11];
    const float* proj_know_b = (const float*)d_in[12];
    const float* tau_know_k  = (const float*)d_in[13];
    const float* tau_know_b  = (const float*)d_in[14];
    const float* expand_O_k  = (const float*)d_in[15];
    const float* n1s = (const float*)d_in[16];
    const float* n1b = (const float*)d_in[17];
    const float* n2s = (const float*)d_in[18];
    const float* n2b = (const float*)d_in[19];

    float* out_x = (float*)d_out;
    float* auxp  = out_x + (out_size - 1);

    // ---- workspace bump allocator ----
    char* base = (char*)d_ws;
    size_t off = 0;
    auto alloc = [&](size_t bytes) -> char* {
        off = (off + 255) & ~(size_t)255;
        char* p = base + off;
        off += bytes;
        return p;
    };
    float* normed   = (float*)alloc((size_t)MROWS * ND * 4);
    bf16*  normed_b = (bf16*) alloc((size_t)MROWS * ND * 2);
    float* h_all    = (float*)alloc((size_t)MROWS * 384 * 4);
    bf16*  h_b      = (bf16*) alloc((size_t)MROWS * DBR * 2);
    float* tauA     = (float*)alloc((size_t)MROWS * 3 * 4);
    float* tauK     = (float*)alloc((size_t)MROWS * 4);
    float* s_raw    = (float*)alloc((size_t)MROWS * N_KNOW * 4);
    bf16*  g_b      = (bf16*) alloc((size_t)MROWS * N_KNOW * 2);
    float* mm_out   = (float*)alloc((size_t)MROWS * ND * 4);
    float* x1       = (float*)alloc((size_t)MROWS * ND * 4);
    bf16*  Qh       = (bf16*) alloc((size_t)NB * NHEADS * NS * DHEAD * 2);
    bf16*  Kh       = (bf16*) alloc((size_t)NB * NHEADS * NS * DHEAD * 2);
    bf16*  VTh      = (bf16*) alloc((size_t)NB * NHEADS * NS * DHEAD * 2);
    bf16*  Ob       = (bf16*) alloc((size_t)MROWS * ND * 2);
    float* cs       = (float*)alloc((size_t)N_KNOW * 4);
    bf16*  projA_T  = (bf16*) alloc((size_t)384 * ND * 2);
    bf16*  qk_wT    = (bf16*) alloc((size_t)ND * N_QK * 2);
    bf16*  v_wT     = (bf16*) alloc((size_t)ND * N_QK * 2);
    bf16*  know_wT  = (bf16*) alloc((size_t)ND * N_KNOW * 2);
    bf16*  expand_T = (bf16*) alloc((size_t)ND * ND * 2);
    bf16*  projK_T  = (bf16*) alloc((size_t)DBR * ND * 2);
    bf16*  qk_nb    = (bf16*) alloc((size_t)N_QK * DBR * 2);
    bf16*  v_nb     = (bf16*) alloc((size_t)N_QK * DBR * 2);
    bf16*  know_nb  = (bf16*) alloc((size_t)N_KNOW * DBR * 2);

    auto eb = [](size_t n) { return (unsigned)((n + 255) / 256); };
    auto gemm = [&](const bf16* A, const bf16* BT, float* C, int M, int N, int K,
                    const float* bias, const float* resid) {
        int blocks = (M / 128) * (N / 128);
        gemm_bf16_wmma<<<blocks, 256, 0, stream>>>(A, BT, C, M, N, K, bias, resid);
    };

    zerof_kernel<<<1, 32, 0, stream>>>(auxp, 1);

    // --- weight preprocessing ---
    transpose_bf16_kernel<<<eb((size_t)ND * 384), 256, 0, stream>>>(proj_attn_k, projA_T, ND, 384);
    transpose_bf16_kernel<<<eb((size_t)N_QK * ND), 256, 0, stream>>>(qk_w,   qk_wT,   N_QK,  ND);
    transpose_bf16_kernel<<<eb((size_t)N_QK * ND), 256, 0, stream>>>(v_w,    v_wT,    N_QK,  ND);
    transpose_bf16_kernel<<<eb((size_t)N_KNOW * ND), 256, 0, stream>>>(know_w, know_wT, N_KNOW, ND);
    transpose_bf16_kernel<<<eb((size_t)ND * ND), 256, 0, stream>>>(expand_O_k, expand_T, ND, ND);
    transpose_bf16_kernel<<<eb((size_t)ND * DBR), 256, 0, stream>>>(proj_know_k, projK_T, ND, DBR);
    rownorm_kernel<<<N_QK, 128, 0, stream>>>(qk_emb,   qk_nb,   DBR);
    rownorm_kernel<<<N_QK, 128, 0, stream>>>(v_emb,    v_nb,    DBR);
    rownorm_kernel<<<N_KNOW, 128, 0, stream>>>(know_emb, know_nb, DBR);

    // --- attention circuit ---
    layernorm_kernel<<<MROWS, 256, 0, stream>>>(x, n1s, n1b, normed, normed_b);
    tau_kernel<<<MROWS, 256, 0, stream>>>(normed, tau_attn_k, tau_attn_b, tauA, 3);
    gemm(normed_b, projA_T, h_all, MROWS, 384, ND, proj_attn_b, nullptr);

    for (int gidx = 0; gidx < 3; ++gidx) {
        slice_bf16_kernel<<<eb((size_t)MROWS * DBR), 256, 0, stream>>>(
            h_all, h_b, MROWS, 384, gidx * DBR, DBR);
        const bf16* emb = (gidx < 2) ? qk_nb : v_nb;
        gemm(h_b, emb, s_raw, MROWS, N_QK, DBR, nullptr, nullptr);
        gate_norm_kernel<<<MROWS, 256, 0, stream>>>(s_raw, tauA + gidx, 3, g_b, N_QK);
        colsum_kernel<<<eb(N_QK), 256, 0, stream>>>(g_b, cs, MROWS, N_QK);
        aux_kernel<<<1, 256, 0, stream>>>(cs, N_QK, 1.0f / (float)N_QK, auxp);
        const bf16* wT = (gidx < 2) ? qk_wT : v_wT;
        gemm(g_b, wT, mm_out, MROWS, ND, N_QK, nullptr, nullptr);
        bf16* dst = (gidx == 0) ? Qh : (gidx == 1) ? Kh : VTh;
        head_prep_kernel<<<eb((size_t)MROWS * ND), 256, 0, stream>>>(mm_out, dst, gidx == 2);
    }

    flash_attn_wmma<<<(NB * NHEADS * (NS / 16)) / 8, 256, 0, stream>>>(Qh, Kh, VTh, mm_out);
    slice_bf16_kernel<<<eb((size_t)MROWS * ND), 256, 0, stream>>>(mm_out, Ob, MROWS, ND, 0, ND);
    gemm(Ob, expand_T, x1, MROWS, ND, ND, nullptr, x);   // x1 = x + attn_out

    // --- knowledge circuit ---
    layernorm_kernel<<<MROWS, 256, 0, stream>>>(x1, n2s, n2b, normed, normed_b);
    tau_kernel<<<MROWS, 256, 0, stream>>>(normed, tau_know_k, tau_know_b, tauK, 1);
    gemm(normed_b, projK_T, h_all, MROWS, DBR, ND, proj_know_b, nullptr);
    slice_bf16_kernel<<<eb((size_t)MROWS * DBR), 256, 0, stream>>>(h_all, h_b, MROWS, DBR, 0, DBR);
    gemm(h_b, know_nb, s_raw, MROWS, N_KNOW, DBR, nullptr, nullptr);
    gate_norm_kernel<<<MROWS, 256, 0, stream>>>(s_raw, tauK, 1, g_b, N_KNOW);
    colsum_kernel<<<eb(N_KNOW), 256, 0, stream>>>(g_b, cs, MROWS, N_KNOW);
    aux_kernel<<<1, 256, 0, stream>>>(cs, N_KNOW, 1.0f / (float)N_KNOW, auxp);
    gemm(g_b, know_wT, out_x, MROWS, ND, N_KNOW, nullptr, x1);  // out = x1 + know_out
}
